// GPT_75136157877068
// MI455X (gfx1250) — matrix operand order repaired and checked by reading, not compiled, not run on hardware
//
#include <hip/hip_runtime.h>
#include <cmath>
#include <cstdint>

// ---------------------------------------------------------------------------
// GPT stack for gfx1250 (MI455X): bf16 WMMA GEMMs + WMMA attention.
// GEMM activation tiles are double-buffered through LDS via the Tensor Data
// Mover (tensor_load_to_lds + s_wait_tensorcnt), overlapping DMA with WMMA.
// B=4, T=1024, C=1024, H=16, Dh=64, FF=4096, L=8.
// ---------------------------------------------------------------------------

#define Bc 4
#define Tc 1024
#define Cc 1024
#define Hc 16
#define Dhc 64
#define FFc 4096
#define Lc 8
#define Mc (Bc * Tc)   // 4096 rows of the activation matrix

typedef __bf16 v16bf __attribute__((ext_vector_type(16)));
typedef __bf16 v8bf  __attribute__((ext_vector_type(8)));
typedef __bf16 v2bf  __attribute__((ext_vector_type(2)));
typedef float  v8f   __attribute__((ext_vector_type(8)));
typedef unsigned int u32;
typedef u32 u32x4 __attribute__((ext_vector_type(4)));
typedef u32 u32x8 __attribute__((ext_vector_type(8)));

// D = A(16x32 bf16) * B(32x16 bf16) + C(16x16 f32)
__device__ __forceinline__ v8f wmma_bf16(v16bf a, v16bf b, v8f c) {
  return __builtin_amdgcn_wmma_f32_16x16x32_bf16(
      /*neg_a=*/false, a, /*neg_b=*/false, b,
      /*c_mod=*/(short)0, c, /*reuse_a=*/false, /*reuse_b=*/false);
}

// Build a 16-element bf16 fragment from two contiguous 8-element runs.
__device__ __forceinline__ v16bf frag_pair(const __bf16* p0, const __bf16* p1) {
  v8bf a = *(const v8bf*)p0;
  v8bf b = *(const v8bf*)p1;
  return __builtin_shufflevector(a, b, 0, 1, 2, 3, 4, 5, 6, 7,
                                       8, 9, 10, 11, 12, 13, 14, 15);
}

// ---------------------------------------------------------------------------
// TDM: DMA a [128 rows x 32 cols] bf16 tile (row stride K elements) from
// global memory into LDS at lds_off, padding each 64B row by 16B so the LDS
// image has a 40-element (80B) row stride (bank-conflict-free b128 reads).
// 2D descriptor: groups 0 and 1 only (VADDR2/3 = NULL form).
// ---------------------------------------------------------------------------
__device__ __forceinline__ void tdm_load_tile_bf16(u32 lds_off,
                                                   const __bf16* gsrc,
                                                   u32 k_elems) {
  const unsigned long long ga = (unsigned long long)(uintptr_t)gsrc;
  u32x4 g0;
  g0[0] = 1u;                                   // count=1, user descriptor
  g0[1] = lds_off;                              // LDS byte address
  g0[2] = (u32)ga;                              // global_addr[31:0]
  g0[3] = (u32)((ga >> 32) & 0x01FFFFFFu) | (2u << 30);  // ga[56:32], type=2
  u32x8 g1;
  // data_size=1 (2B) | pad_enable | pad_interval=3 (16 DW) | pad_amount=3 (4 DW)
  g1[0] = (1u << 16) | (1u << 20) | (3u << 22) | (3u << 25);
  g1[1] = (k_elems & 0xFFFFu) << 16;            // tensor_dim0[15:0]
  g1[2] = (k_elems >> 16) & 0xFFFFu;            // tensor_dim0[31:16], td1 lo=0
  g1[3] = 16u | (32u << 16);                    // tensor_dim1=1<<20, tile_dim0=32
  g1[4] = 128u;                                 // tile_dim1=128, tile_dim2=0
  g1[5] = k_elems;                              // tensor_dim0_stride[31:0]
  g1[6] = 0u;                                   // stride0 hi, stride1 lo
  g1[7] = 0u;
  asm volatile("tensor_load_to_lds %0, %1" : : "s"(g0), "s"(g1) : "memory");
}

// ---------------------------------------------------------------------------
// LayerNorm: fp32 row -> normalized bf16 row. One block per row.
// ---------------------------------------------------------------------------
__global__ __launch_bounds__(256) void ln_kernel(const float* __restrict__ x,
                                                 const float* __restrict__ w,
                                                 const float* __restrict__ b,
                                                 __bf16* __restrict__ out) {
  const int row = blockIdx.x;
  const int tid = threadIdx.x;
  const float* xr = x + (size_t)row * Cc;

  float s = 0.f, s2 = 0.f;
  for (int i = tid; i < Cc; i += 256) {
    float v = xr[i];
    s += v;
    s2 += v * v;
  }
  __shared__ float r1[256];
  __shared__ float r2[256];
  r1[tid] = s;
  r2[tid] = s2;
  __syncthreads();
  for (int off = 128; off > 0; off >>= 1) {
    if (tid < off) {
      r1[tid] += r1[tid + off];
      r2[tid] += r2[tid + off];
    }
    __syncthreads();
  }
  const float mu = r1[0] * (1.f / Cc);
  const float var = r2[0] * (1.f / Cc) - mu * mu;
  const float rs = rsqrtf(var + 1e-5f);
  __bf16* orow = out + (size_t)row * Cc;
  for (int i = tid; i < Cc; i += 256) {
    orow[i] = (__bf16)((xr[i] - mu) * rs * w[i] + b[i]);
  }
}

// ---------------------------------------------------------------------------
// GEMM: Y[M,N] = X[M,K](bf16) @ W[K,N](f32, converted to bf16) + bias.
// Block = 256 threads (8 waves, 4x2), tile 128x128, K-step 32.
// Double-buffered: TDM prefetches the next X tile and VALU stages the next W
// tile while WMMA consumes the current buffers; one barrier per k-step.
// EPI 0: bias -> bf16 out
// EPI 1: bias + residual(f32) -> f32 out
// EPI 2: bias + exact GELU -> bf16 out
// ---------------------------------------------------------------------------
template <int EPI>
__global__ __launch_bounds__(256) void gemm_kernel(
    const __bf16* __restrict__ X, const float* __restrict__ W,
    const float* __restrict__ bias, void* __restrict__ Yv,
    const float* __restrict__ resid, int M, int N, int K) {
  __shared__ __bf16 Xs[2][128][40];  // [m][k], 80B stride (conflict-free b128)
  __shared__ __bf16 Ws[2][128][40];  // [n][k] (W transposed tile)

  const int tid = threadIdx.x;
  const int lane = tid & 31;
  const int wave = tid >> 5;
  const int hl = lane & 15;
  const int hi = (lane >> 4) & 1;
  const int wm = wave >> 1;  // 0..3
  const int wn = wave & 1;   // 0..1
  const int n0 = blockIdx.x * 128;
  const int m0 = blockIdx.y * 128;
  const u32 xsLds[2] = {(u32)(uintptr_t)&Xs[0][0][0],
                        (u32)(uintptr_t)&Xs[1][0][0]};

  // Stage W tile transposed with f32->bf16 conversion: 32x128.
  // Each thread: 2 k-rows x 8 n, packed bf16-pair LDS stores.
  const int kp2 = tid >> 4;       // 0..15 -> k = 2*kp2
  const int ng = (tid & 15) * 8;  // 0..120
  auto stage_w = [&](int buf, int ksw) {
    const float* w0p = W + (size_t)(ksw + 2 * kp2) * N + n0 + ng;
    const float* w1p = w0p + N;
#pragma unroll
    for (int j = 0; j < 8; ++j) {
      v2bf pk = {(__bf16)w0p[j], (__bf16)w1p[j]};
      *(v2bf*)&Ws[buf][ng + j][2 * kp2] = pk;
    }
  };

  v8f acc[2][4] = {};

  // Prologue: stage k-step 0 into buffer 0.
  if (wave == 0) tdm_load_tile_bf16(xsLds[0], X + (size_t)m0 * K, (u32)K);
  stage_w(0, 0);
  if (wave == 0) __builtin_amdgcn_s_wait_tensorcnt(0);
  __syncthreads();

  for (int ks = 0; ks < K; ks += 32) {
    const int cur = (ks >> 5) & 1;
    const int nxt = cur ^ 1;
    // Prefetch next k-step into the other buffer (overlaps WMMA below).
    if (ks + 32 < K) {
      if (wave == 0)
        tdm_load_tile_bf16(xsLds[nxt], X + (size_t)m0 * K + ks + 32, (u32)K);
      stage_w(nxt, ks + 32);
    }

    v16bf af[2], bw[4];
#pragma unroll
    for (int tm = 0; tm < 2; ++tm) {
      const __bf16* p = &Xs[cur][wm * 32 + tm * 16 + hl][0];
      af[tm] = frag_pair(p + 8 * hi, p + 16 + 8 * hi);  // A layout
    }
#pragma unroll
    for (int tn = 0; tn < 4; ++tn) {
      const __bf16* p = &Ws[cur][wn * 64 + tn * 16 + hl][0];
      bw[tn] = frag_pair(p + 8 * hi, p + 16 + 8 * hi);  // B layout
    }
#pragma unroll
    for (int tm = 0; tm < 2; ++tm)
#pragma unroll
      for (int tn = 0; tn < 4; ++tn)
        acc[tm][tn] = wmma_bf16(af[tm], bw[tn], acc[tm][tn]);

    // Publish the prefetched buffers; also fences reads of `cur`.
    if (wave == 0) __builtin_amdgcn_s_wait_tensorcnt(0);
    __syncthreads();
  }

  // Epilogue: c[r] holds element (m = r + 8*hi, n = halfLane) of each tile.
#pragma unroll
  for (int tm = 0; tm < 2; ++tm) {
#pragma unroll
    for (int tn = 0; tn < 4; ++tn) {
      const int nn = n0 + wn * 64 + tn * 16 + hl;
      const float bv = bias[nn];
#pragma unroll
      for (int r = 0; r < 8; ++r) {
        const int mm = m0 + wm * 32 + tm * 16 + r + 8 * hi;
        float v = acc[tm][tn][r] + bv;
        const size_t idx = (size_t)mm * N + nn;
        if (EPI == 0) {
          ((__bf16*)Yv)[idx] = (__bf16)v;
        } else if (EPI == 1) {
          ((float*)Yv)[idx] = v + resid[idx];
        } else {
          v = 0.5f * v * (1.0f + erff(v * 0.70710678118654752f));
          ((__bf16*)Yv)[idx] = (__bf16)v;
        }
      }
    }
  }
}

// ---------------------------------------------------------------------------
// Attention: one block per (b, h, 64-query tile). 256 threads, 8 waves.
// Dynamic LDS: S[64][1024] f32 + rowmax[64] + rowsum[64] + Vt[64][40] bf16.
// Mask computed analytically: keep = j<=i || (i<T-1 && i/10==j/10 && j%10<=8).
// ---------------------------------------------------------------------------
__global__ __launch_bounds__(256) void attn_kernel(
    const __bf16* __restrict__ Q, const __bf16* __restrict__ Kq,
    const __bf16* __restrict__ V, __bf16* __restrict__ Y) {
  extern __shared__ float smem[];
  float* S = smem;                          // 64 * 1024 f32
  float* rowmax = smem + 64 * 1024;         // 64
  float* rowsum = rowmax + 64;              // 64
  __bf16* Vt = (__bf16*)(rowsum + 64);      // [64][40] bf16 (d-major)

  const int qb = blockIdx.x;
  const int h = blockIdx.y;
  const int b = blockIdx.z;
  const int tid = threadIdx.x;
  const int lane = tid & 31;
  const int wave = tid >> 5;
  const int hl = lane & 15;
  const int hi = (lane >> 4) & 1;

  const int q0 = qb * 64;
  // tokenlearner extra mask reaches at most j <= i + 8; round K range to 32.
  const int kEnd = ((q0 + 96) < Tc) ? (q0 + 96) : Tc;
  const float scale = 0.125f;  // 1/sqrt(64)

  // ---- Phase 1: S = mask(QK^T * scale) --------------------------------
  {
    const int qt = wave & 3;   // query 16-tile within the 64 block
    const int kh = wave >> 2;  // key half
    const __bf16* qrow =
        Q + ((size_t)(b * Tc + q0 + qt * 16 + hl) * Cc + h * Dhc);
    v16bf aQ[2];
#pragma unroll
    for (int kk = 0; kk < 2; ++kk)
      aQ[kk] = frag_pair(qrow + kk * 32 + 8 * hi, qrow + kk * 32 + 16 + 8 * hi);

    const int kbeg = kh * 512;
    const int klim = (kEnd < kbeg + 512) ? kEnd : (kbeg + 512);
    for (int k0 = kbeg; k0 < klim; k0 += 16) {
      const __bf16* krow =
          Kq + ((size_t)(b * Tc + k0 + hl) * Cc + h * Dhc);  // lane = key col
      v16bf bK0 = frag_pair(krow + 8 * hi, krow + 16 + 8 * hi);
      v16bf bK1 = frag_pair(krow + 32 + 8 * hi, krow + 48 + 8 * hi);
      v8f c = {};
      c = wmma_bf16(aQ[0], bK0, c);
      c = wmma_bf16(aQ[1], bK1, c);
      const int j = k0 + hl;  // key index for this lane
#pragma unroll
      for (int r = 0; r < 8; ++r) {
        const int ml = qt * 16 + r + 8 * hi;
        const int i = q0 + ml;  // query index
        const bool keep =
            (j <= i) || ((i < Tc - 1) && (i / 10 == j / 10) && (j % 10 <= 8));
        S[ml * Tc + j] = keep ? c[r] * scale : -1e30f;
      }
    }
  }
  __syncthreads();

  // ---- Phase 2: per-row max and sum(exp) ------------------------------
  {
    const int row = tid >> 2;
    const int part = tid & 3;
    const float* Sr = S + row * Tc;
    float mx = -1e30f;
    for (int j = part; j < kEnd; j += 4) mx = fmaxf(mx, Sr[j]);
    mx = fmaxf(mx, __shfl_xor(mx, 1, 32));
    mx = fmaxf(mx, __shfl_xor(mx, 2, 32));
    float sm = 0.f;
    for (int j = part; j < kEnd; j += 4) sm += __expf(Sr[j] - mx);
    sm += __shfl_xor(sm, 1, 32);
    sm += __shfl_xor(sm, 2, 32);
    if (part == 0) {
      rowmax[row] = mx;
      rowsum[row] = sm;
    }
  }
  __syncthreads();

  // ---- Phase 3: O = softmax(S) @ V via WMMA ---------------------------
  // kEnd is block-uniform, so all waves share the k-loop and a staged
  // transposed V tile in LDS.
  {
    const int qt = wave & 3;
    const int nh = wave >> 2;  // which 32-wide half of Dh
    v8f o[2] = {};
    const int mlA = qt * 16 + hl;  // A-fragment row for this lane
    const float rm = rowmax[mlA];

    for (int k0 = 0; k0 < kEnd; k0 += 32) {
      // Stage Vt[d][k] = V[k0+k][h*64+d]: vectorized 16B reads, transpose.
      {
        const int kk = tid >> 3;        // 0..31
        const int dg = (tid & 7) * 8;   // 0..56
        const __bf16* vp =
            V + (size_t)(b * Tc + k0 + kk) * Cc + h * Dhc + dg;
        v8bf vv = *(const v8bf*)vp;
#pragma unroll
        for (int jj = 0; jj < 8; ++jj) Vt[(dg + jj) * 40 + kk] = vv[jj];
      }
      __syncthreads();

      // A fragment of P = exp(S - rowmax), built in registers.
      const float* s0 = S + mlA * Tc + k0 + 8 * hi;
      const float* s1 = s0 + 16;
      v16bf aP;
#pragma unroll
      for (int e = 0; e < 8; ++e) aP[e] = (__bf16)__expf(s0[e] - rm);
#pragma unroll
      for (int e = 0; e < 8; ++e) aP[8 + e] = (__bf16)__expf(s1[e] - rm);

#pragma unroll
      for (int nt = 0; nt < 2; ++nt) {
        const int dl = nh * 32 + nt * 16 + hl;  // lane = local d column
        const __bf16* p = &Vt[dl * 40 + 8 * hi];
        v16bf bV = frag_pair(p, p + 16);
        o[nt] = wmma_bf16(aP, bV, o[nt]);
      }
      __syncthreads();
    }

#pragma unroll
    for (int nt = 0; nt < 2; ++nt) {
#pragma unroll
      for (int r = 0; r < 8; ++r) {
        const int ml = qt * 16 + r + 8 * hi;
        const float val = o[nt][r] / rowsum[ml];
        Y[(size_t)(b * Tc + q0 + ml) * Cc + h * Dhc + nh * 32 + nt * 16 + hl] =
            (__bf16)val;
      }
    }
  }
}

// ---------------------------------------------------------------------------
// Host launcher
// ---------------------------------------------------------------------------
extern "C" void kernel_launch(void* const* d_in, const int* in_sizes, int n_in,
                              void* d_out, int out_size, void* d_ws,
                              size_t ws_size, hipStream_t stream) {
  const float* x = (const float*)d_in[0];
  const float* ln1w = (const float*)d_in[1];
  const float* ln1b = (const float*)d_in[2];
  const float* ln2w = (const float*)d_in[3];
  const float* ln2b = (const float*)d_in[4];
  const float* wq = (const float*)d_in[5];
  const float* bq = (const float*)d_in[6];
  const float* wk = (const float*)d_in[7];
  const float* bk = (const float*)d_in[8];
  const float* wv = (const float*)d_in[9];
  const float* bv = (const float*)d_in[10];
  const float* wo = (const float*)d_in[11];
  const float* bo = (const float*)d_in[12];
  const float* w1 = (const float*)d_in[13];
  const float* b1 = (const float*)d_in[14];
  const float* w2 = (const float*)d_in[15];
  const float* b2 = (const float*)d_in[16];
  // d_in[17] = mask: computed analytically on device.
  float* xout = (float*)d_out;

  char* ws = (char*)d_ws;
  __bf16* hbuf = (__bf16*)(ws);                         // M*C bf16 (8 MB)
  __bf16* qbuf = (__bf16*)(ws + ((size_t)8 << 20));     // 8 MB
  __bf16* kbuf = (__bf16*)(ws + ((size_t)16 << 20));    // 8 MB
  __bf16* vbuf = (__bf16*)(ws + ((size_t)24 << 20));    // 8 MB
  __bf16* ybuf = (__bf16*)(ws + ((size_t)32 << 20));    // 8 MB
  __bf16* mbuf = (__bf16*)(ws + ((size_t)40 << 20));    // M*FF bf16 (32 MB)

  // Residual stream lives in d_out (fp32), updated in place.
  hipMemcpyAsync(xout, x, (size_t)Mc * Cc * sizeof(float),
                 hipMemcpyDeviceToDevice, stream);

  const dim3 gC(Cc / 128, Mc / 128);
  const dim3 gF(FFc / 128, Mc / 128);
  const dim3 gA(Tc / 64, Hc, Bc);
  const size_t attn_smem =
      (size_t)(64 * 1024 + 128) * sizeof(float) + (size_t)64 * 40 * 2;

  for (int l = 0; l < Lc; ++l) {
    ln_kernel<<<Mc, 256, 0, stream>>>(xout, ln1w + (size_t)l * Cc,
                                      ln1b + (size_t)l * Cc, hbuf);
    gemm_kernel<0><<<gC, 256, 0, stream>>>(hbuf, wq + (size_t)l * Cc * Cc,
                                           bq + (size_t)l * Cc, qbuf, nullptr,
                                           Mc, Cc, Cc);
    gemm_kernel<0><<<gC, 256, 0, stream>>>(hbuf, wk + (size_t)l * Cc * Cc,
                                           bk + (size_t)l * Cc, kbuf, nullptr,
                                           Mc, Cc, Cc);
    gemm_kernel<0><<<gC, 256, 0, stream>>>(hbuf, wv + (size_t)l * Cc * Cc,
                                           bv + (size_t)l * Cc, vbuf, nullptr,
                                           Mc, Cc, Cc);
    attn_kernel<<<gA, 256, attn_smem, stream>>>(qbuf, kbuf, vbuf, ybuf);
    gemm_kernel<1><<<gC, 256, 0, stream>>>(ybuf, wo + (size_t)l * Cc * Cc,
                                           bo + (size_t)l * Cc, xout, xout, Mc,
                                           Cc, Cc);
    ln_kernel<<<Mc, 256, 0, stream>>>(xout, ln2w + (size_t)l * Cc,
                                      ln2b + (size_t)l * Cc, hbuf);
    gemm_kernel<2><<<gF, 256, 0, stream>>>(hbuf, w1 + (size_t)l * Cc * FFc,
                                           b1 + (size_t)l * FFc, mbuf, nullptr,
                                           Mc, FFc, Cc);
    gemm_kernel<1><<<gC, 256, 0, stream>>>(mbuf, w2 + (size_t)l * FFc * Cc,
                                           b2 + (size_t)l * Cc, xout, xout, Mc,
                                           Cc, FFc);
  }
}
